// ResBlock_21105469292593
// MI455X (gfx1250) — matrix operand order repaired and checked by reading, not compiled, not run on hardware
//
#include <hip/hip_runtime.h>
#include <hip/hip_bf16.h>
#include <stdint.h>

typedef __attribute__((ext_vector_type(16))) _Float16 v16h;
typedef __attribute__((ext_vector_type(8)))  _Float16 v8h;
typedef __attribute__((ext_vector_type(8)))  float    v8f;

#define CIN   384
#define COUT  512
#define HWP   3136      // 56*56
#define NPIX  50176     // 16*3136
#define LDA   392       // padded f16 LDS row stride (multiple of 8 -> 16B aligned)
#define MB    64        // pixels (M) per block; 3136 % 64 == 0 -> no batch crossing

// one-instruction XOR-lane shuffle (all lanes active): ds_swizzle_b32,
// group-of-32 mode: and=0x1F, or=0, xor=mask  ->  offset = 0x1F | mask<<10
#define XORSWZ(x, mask) \
    __builtin_bit_cast(float, __builtin_amdgcn_ds_swizzle( \
        __builtin_bit_cast(int, (x)), (0x1f | ((mask) << 10))))

// ---------------- kernel 0: f32 -> f16 weight conversion ----------------
__global__ void k0_cvt(const float* __restrict__ w, _Float16* __restrict__ o, int n) {
    int i = blockIdx.x * blockDim.x + threadIdx.x;
    if (i < n) o[i] = (_Float16)w[i];
}

// ---------------- kernel 1: dual LayerNorm on NCHW x --------------------
__global__ __launch_bounds__(256) void k1_ln(
        const float* __restrict__ x,
        const float* __restrict__ g1, const float* __restrict__ b1,
        const float* __restrict__ gs, const float* __restrict__ bsl,
        _Float16* __restrict__ A1, _Float16* __restrict__ AS) {
    int p = blockIdx.x * blockDim.x + threadIdx.x;
    if (p >= NPIX) return;
    int b = p / HWP, hw = p % HWP;
    const float* xb = x + (size_t)b * CIN * HWP + hw;
    float s = 0.f, ss = 0.f;
    for (int c = 0; c < CIN; ++c) {
        float v = xb[(size_t)c * HWP];
        s += v; ss += v * v;
    }
    float mu   = s * (1.0f / CIN);
    float var  = ss * (1.0f / CIN) - mu * mu;
    float rstd = rsqrtf(var + 1e-5f);
    _Float16* a1 = A1 + (size_t)p * CIN;
    _Float16* as = AS + (size_t)p * CIN;
    for (int c = 0; c < CIN; ++c) {
        float v = (xb[(size_t)c * HWP] - mu) * rstd;   // 2nd pass hits L2 (x=77MB < 192MB)
        a1[c] = (_Float16)fmaxf(v * g1[c] + b1[c], 0.f);
        as[c] = (_Float16)(v * gs[c] + bsl[c]);
    }
}

// ---- fragment loaders (ISA 7.12.2 16-bit layouts; half = lane>>4) ------
__device__ inline v16h fragA_lds(const _Float16* row, int k0, int half) {
    union { v16h v; v8h h[2]; } u;
    u.h[0] = *(const v8h*)(row + k0 +      half * 8);
    u.h[1] = *(const v8h*)(row + k0 + 16 + half * 8);
    return u.v;
}
__device__ inline v16h fragA_gbl(const _Float16* __restrict__ A, int prow, int k0, int half) {
    union { v16h v; v8h h[2]; } u;
    const _Float16* r = A + (size_t)prow * CIN + k0;
    u.h[0] = *(const v8h*)(r +      half * 8);
    u.h[1] = *(const v8h*)(r + 16 + half * 8);
    return u.v;
}
__device__ inline v16h fragB_gbl(const _Float16* __restrict__ W, int n, int k0, int half) {
    union { v16h v; v8h h[2]; } u;
    const _Float16* r = W + (size_t)n * CIN + k0;
    u.h[0] = *(const v8h*)(r +      half * 8);
    u.h[1] = *(const v8h*)(r + 16 + half * 8);
    return u.v;
}

// ---------------- kernel 2: GEMM1 (384->384) + LN2 + ReLU ---------------
// block = 64 pixels x 384 outputs; wave: 4 M-tiles x 3 N-tiles (12 WMMA/kstep)
// weights read ONCE per block; LN2 stats from registers (no f32 C staging)
__global__ __launch_bounds__(256) void k2_gemm1(
        const _Float16* A1, const _Float16* __restrict__ W1h,
        const float* __restrict__ g2, const float* __restrict__ b2ln,
        _Float16* Hout) {
    __shared__ _Float16 As[MB * LDA];      // 50,176 B
    __shared__ float Ps [MB * 8];          // per-wave partial sums
    __shared__ float Pss[MB * 8];
    __shared__ float Mu [MB];
    __shared__ float Rsd[MB];
    const int tid  = threadIdx.x;
    const int lane = tid & 31;
    const int wav  = tid >> 5;
    const int half = lane >> 4;
    const int ml   = lane & 15;
    const int p0   = blockIdx.x * MB;

    for (int q = tid; q < MB * 96; q += 256) {
        int r = q / 96, cc = (q % 96) * 4;
        *(uint2*)(&As[r * LDA + cc]) = *(const uint2*)(&A1[(size_t)(p0 + r) * CIN + cc]);
    }
    __syncthreads();

    v8f acc[4][3];
    #pragma unroll
    for (int mt = 0; mt < 4; ++mt)
        #pragma unroll
        for (int t = 0; t < 3; ++t)
            acc[mt][t] = (v8f){0.f,0.f,0.f,0.f,0.f,0.f,0.f,0.f};

    for (int ks = 0; ks < 12; ++ks) {
        const int k0 = ks * 32;
        v16h a[4];
        #pragma unroll
        for (int mt = 0; mt < 4; ++mt)
            a[mt] = fragA_lds(&As[(mt * 16 + ml) * LDA], k0, half);
        #pragma unroll
        for (int t = 0; t < 3; ++t) {
            v16h bm = fragB_gbl(W1h, wav * 48 + t * 16 + ml, k0, half);
            #pragma unroll
            for (int mt = 0; mt < 4; ++mt)
                acc[mt][t] = __builtin_amdgcn_wmma_f32_16x16x32_f16(
                                false, a[mt], false, bm, (short)0, acc[mt][t], false, false);
        }
    }

    // per-row partials over this wave's 48 columns (XOR reduce within 16-lane half)
    #pragma unroll
    for (int mt = 0; mt < 4; ++mt) {
        #pragma unroll
        for (int r = 0; r < 8; ++r) {
            float v0 = acc[mt][0][r], v1 = acc[mt][1][r], v2 = acc[mt][2][r];
            float s  = v0 + v1 + v2;
            float ss = v0*v0 + v1*v1 + v2*v2;
            s += XORSWZ(s, 8);  ss += XORSWZ(ss, 8);
            s += XORSWZ(s, 4);  ss += XORSWZ(ss, 4);
            s += XORSWZ(s, 2);  ss += XORSWZ(ss, 2);
            s += XORSWZ(s, 1);  ss += XORSWZ(ss, 1);
            if (ml == 0) {
                int mrow = mt * 16 + r + 8 * half;
                Ps [mrow * 8 + wav] = s;
                Pss[mrow * 8 + wav] = ss;
            }
        }
    }
    __syncthreads();
    if (tid < MB) {
        float s = 0.f, ss = 0.f;
        #pragma unroll
        for (int w = 0; w < 8; ++w) { s += Ps[tid * 8 + w]; ss += Pss[tid * 8 + w]; }
        float mu  = s * (1.0f / CIN);
        float var = ss * (1.0f / CIN) - mu * mu;
        Mu[tid]  = mu;
        Rsd[tid] = rsqrtf(var + 1e-5f);
    }
    __syncthreads();

    // normalize + ReLU + f16 store straight from registers
    #pragma unroll
    for (int t = 0; t < 3; ++t) {
        const int d = wav * 48 + t * 16 + ml;
        const float gg = g2[d], bb = b2ln[d];
        #pragma unroll
        for (int mt = 0; mt < 4; ++mt) {
            #pragma unroll
            for (int r = 0; r < 8; ++r) {
                int mrow = mt * 16 + r + 8 * half;
                float hn = fmaxf((acc[mt][t][r] - Mu[mrow]) * Rsd[mrow] * gg + bb, 0.f);
                Hout[(size_t)(p0 + mrow) * CIN + d] = (_Float16)hn;
            }
        }
    }
}

// ---------------- kernel 3: GEMM2 + shortcut GEMM + bias + NCHW out ----
// block = 64 pixels x 512 outputs; wave: 4 M-tiles x 4 N-tiles (16 WMMA/kstep)
// C VGPR layout = 8 consecutive pixels per lane at fixed channel -> direct
// contiguous v8f stores into NCHW (no output staging)
__global__ __launch_bounds__(256) void k3_gemm2(
        const _Float16* __restrict__ Hin, const _Float16* __restrict__ ASin,
        const _Float16* __restrict__ W2h, const _Float16* __restrict__ WSh,
        const float* __restrict__ b2, const float* __restrict__ bs,
        float* __restrict__ out) {
    __shared__ _Float16 Hs[MB * LDA];      // 50,176 B (only main-path A staged)
    const int tid  = threadIdx.x;
    const int lane = tid & 31;
    const int wav  = tid >> 5;
    const int half = lane >> 4;
    const int ml   = lane & 15;
    const int p0   = blockIdx.x * MB;

    for (int q = tid; q < MB * 96; q += 256) {
        int r = q / 96, cc = (q % 96) * 4;
        *(uint2*)(&Hs[r * LDA + cc]) = *(const uint2*)(&Hin[(size_t)(p0 + r) * CIN + cc]);
    }
    __syncthreads();

    v8f acc[4][4];
    #pragma unroll
    for (int mt = 0; mt < 4; ++mt)
        #pragma unroll
        for (int t = 0; t < 4; ++t)
            acc[mt][t] = (v8f){0.f,0.f,0.f,0.f,0.f,0.f,0.f,0.f};

    for (int ks = 0; ks < 12; ++ks) {       // main path: H @ w2^T (A from LDS)
        const int k0 = ks * 32;
        v16h a[4];
        #pragma unroll
        for (int mt = 0; mt < 4; ++mt)
            a[mt] = fragA_lds(&Hs[(mt * 16 + ml) * LDA], k0, half);
        #pragma unroll
        for (int t = 0; t < 4; ++t) {
            v16h bm = fragB_gbl(W2h, wav * 64 + t * 16 + ml, k0, half);
            #pragma unroll
            for (int mt = 0; mt < 4; ++mt)
                acc[mt][t] = __builtin_amdgcn_wmma_f32_16x16x32_f16(
                                false, a[mt], false, bm, (short)0, acc[mt][t], false, false);
        }
    }
    for (int ks = 0; ks < 12; ++ks) {       // shortcut: LNs(x) @ ws^T (A from global/L2)
        const int k0 = ks * 32;
        v16h a[4];
        #pragma unroll
        for (int mt = 0; mt < 4; ++mt)
            a[mt] = fragA_gbl(ASin, p0 + mt * 16 + ml, k0, half);
        #pragma unroll
        for (int t = 0; t < 4; ++t) {
            v16h bm = fragB_gbl(WSh, wav * 64 + t * 16 + ml, k0, half);
            #pragma unroll
            for (int mt = 0; mt < 4; ++mt)
                acc[mt][t] = __builtin_amdgcn_wmma_f32_16x16x32_f16(
                                false, a[mt], false, bm, (short)0, acc[mt][t], false, false);
        }
    }

    // bias + direct NCHW stores: lane holds 8 consecutive pixels @ fixed d
    const int b  = p0 / HWP;
    const int hw = p0 % HWP;                 // 64-aligned
    #pragma unroll
    for (int t = 0; t < 4; ++t) {
        const int d = wav * 64 + t * 16 + ml;
        const float bias = b2[d] + bs[d];
        const size_t base = ((size_t)b * COUT + d) * HWP + hw + 8 * half;
        #pragma unroll
        for (int mt = 0; mt < 4; ++mt) {
            v8f vv;
            #pragma unroll
            for (int r = 0; r < 8; ++r) vv[r] = acc[mt][t][r] + bias;
            *(v8f*)(&out[base + mt * 16]) = vv;   // 32B contiguous; l & l+16 fill 64B line
        }
    }
}

// ---------------- launcher ---------------------------------------------
extern "C" void kernel_launch(void* const* d_in, const int* in_sizes, int n_in,
                              void* d_out, int out_size, void* d_ws, size_t ws_size,
                              hipStream_t stream) {
    const float* x    = (const float*)d_in[0];
    const float* g1   = (const float*)d_in[1];
    const float* b1   = (const float*)d_in[2];
    const float* w1   = (const float*)d_in[3];
    const float* g2   = (const float*)d_in[4];
    const float* b2ln = (const float*)d_in[5];
    const float* w2   = (const float*)d_in[6];
    const float* b2   = (const float*)d_in[7];
    const float* gs   = (const float*)d_in[8];
    const float* bsl  = (const float*)d_in[9];
    const float* wsc  = (const float*)d_in[10];
    const float* bsc  = (const float*)d_in[11];
    float* out = (float*)d_out;

    char* wsb = (char*)d_ws;
    const size_t SZ_ACT = (size_t)NPIX * CIN * sizeof(_Float16);   // 38,535,168
    _Float16* A1  = (_Float16*)(wsb);                              // aliased as H after k2
    _Float16* AS  = (_Float16*)(wsb + SZ_ACT);
    _Float16* W1h = (_Float16*)(wsb + 2 * SZ_ACT);
    _Float16* W2h = (_Float16*)(wsb + 2 * SZ_ACT + (size_t)CIN * CIN * 2);
    _Float16* WSh = (_Float16*)(wsb + 2 * SZ_ACT + (size_t)CIN * CIN * 2
                                               + (size_t)COUT * CIN * 2);

    k0_cvt<<<(CIN * CIN  + 255) / 256, 256, 0, stream>>>(w1,  W1h, CIN * CIN);
    k0_cvt<<<(COUT * CIN + 255) / 256, 256, 0, stream>>>(w2,  W2h, COUT * CIN);
    k0_cvt<<<(COUT * CIN + 255) / 256, 256, 0, stream>>>(wsc, WSh, COUT * CIN);

    k1_ln<<<(NPIX + 255) / 256, 256, 0, stream>>>(x, g1, b1, gs, bsl, A1, AS);
    k2_gemm1<<<NPIX / MB, 256, 0, stream>>>(A1, W1h, g2, b2ln, A1 /*Hout in place*/);
    k3_gemm2<<<NPIX / MB, 256, 0, stream>>>(A1, AS, W2h, WSh, b2, bsc, out);
}
